// One2ManyAttention_8031588843668
// MI455X (gfx1250) — matrix operand order repaired and checked by reading, not compiled, not run on hardware
//
#include <hip/hip_runtime.h>
#include <hip/hip_bf16.h>
#include <math.h>

#define NHEAD 8
#define DIM 16
#define EPS_F 1e-6f

typedef __attribute__((ext_vector_type(16))) __bf16 v16bf;
typedef __attribute__((ext_vector_type(8)))  __bf16 v8bf;
typedef __attribute__((ext_vector_type(8)))  float  v8f;

__device__ __forceinline__ v8bf zero8bf() {
    v8bf z;
    #pragma unroll
    for (int j = 0; j < 8; ++j) z[j] = (__bf16)0.0f;
    return z;
}

__device__ __forceinline__ v16bf zero16bf() {
    v16bf z;
    #pragma unroll
    for (int j = 0; j < 16; ++j) z[j] = (__bf16)0.0f;
    return z;
}

__device__ __forceinline__ float featmap(float x) {
    // elu(x) + 1
    return x > 0.0f ? x + 1.0f : __expf(x);
}

__device__ __forceinline__ void inv3(const float m[9], float o[9]) {
    float A =  (m[4]*m[8] - m[5]*m[7]);
    float B = -(m[3]*m[8] - m[5]*m[6]);
    float C =  (m[3]*m[7] - m[4]*m[6]);
    float det = m[0]*A + m[1]*B + m[2]*C;
    float inv = 1.0f / det;
    o[0] = A*inv;
    o[1] = -(m[1]*m[8] - m[2]*m[7])*inv;
    o[2] =  (m[1]*m[5] - m[2]*m[4])*inv;
    o[3] = B*inv;
    o[4] =  (m[0]*m[8] - m[2]*m[6])*inv;
    o[5] = -(m[0]*m[5] - m[2]*m[3])*inv;
    o[6] = C*inv;
    o[7] = -(m[0]*m[7] - m[1]*m[6])*inv;
    o[8] =  (m[0]*m[4] - m[1]*m[3])*inv;
}

// One-shot: phi(Q), phi(K) -> bf16 (same [n][l][h][d] layout); V -> bf16 transposed [n][h][v][s].
__global__ void prep_kernel(const float* __restrict__ Q,
                            const float* __restrict__ K,
                            const float* __restrict__ V,
                            __bf16* __restrict__ Qb,
                            __bf16* __restrict__ Kb,
                            __bf16* __restrict__ Vt,
                            int N, int L) {
    size_t idx = (size_t)blockIdx.x * blockDim.x + threadIdx.x;
    size_t total = (size_t)N * L * NHEAD * DIM;
    if (idx >= total) return;
    int v = (int)(idx % DIM);
    int h = (int)((idx / DIM) % NHEAD);
    size_t l = (idx / (DIM * NHEAD)) % (size_t)L;
    size_t n = idx / ((size_t)DIM * NHEAD * L);
    Qb[idx] = (__bf16)featmap(Q[idx]);
    Kb[idx] = (__bf16)featmap(K[idx]);
    Vt[((n * NHEAD + h) * DIM + v) * (size_t)L + l] = (__bf16)V[idx];
}

// Per (n, l) epipolar line params: center = p * par_coord + q ; mode flag.
// Also writes the shared per-key coordinate table coords[l] = (x, y).
__global__ void epi_lines_kernel(const float* __restrict__ K0g,
                                 const float* __restrict__ K1g,
                                 const float* __restrict__ Rg,
                                 const float* __restrict__ tg,
                                 const int* __restrict__ w0c_p,
                                 const int* __restrict__ scale_p,
                                 int N, int L,
                                 float4* __restrict__ lines,
                                 float2* __restrict__ coords) {
    int idx = blockIdx.x * blockDim.x + threadIdx.x;
    if (idx >= N * L) return;
    int n = idx / L;
    int l = idx - n * L;
    int w0c = *w0c_p;
    float invs = 1.0f / (float)(*scale_p);

    float x = (float)(l % w0c);
    float y = (float)(l / w0c);
    if (n == 0) coords[l] = make_float2(x, y);

    float K0s[9], K1s[9];
    #pragma unroll
    for (int i = 0; i < 9; ++i) { K0s[i] = K0g[n*9+i]; K1s[i] = K1g[n*9+i]; }
    #pragma unroll
    for (int i = 0; i < 6; ++i) { K0s[i] *= invs; K1s[i] *= invs; } // scale rows 0,1

    float tx = tg[n*3+0], ty = tg[n*3+1], tz = tg[n*3+2];
    float S[9] = {0.f,-tz,ty,  tz,0.f,-tx,  -ty,tx,0.f};
    float E[9];
    #pragma unroll
    for (int i = 0; i < 3; ++i)
        #pragma unroll
        for (int j = 0; j < 3; ++j) {
            float s = 0.f;
            #pragma unroll
            for (int k = 0; k < 3; ++k) s += S[i*3+k] * Rg[n*9 + k*3 + j];
            E[i*3+j] = s;
        }
    float iK0[9], iK1[9];
    inv3(K0s, iK0); inv3(K1s, iK1);
    float T[9];
    #pragma unroll
    for (int i = 0; i < 3; ++i)
        #pragma unroll
        for (int j = 0; j < 3; ++j) {
            float s = 0.f;
            #pragma unroll
            for (int k = 0; k < 3; ++k) s += E[i*3+k] * iK0[k*3+j];
            T[i*3+j] = s;
        }
    float F[9];
    #pragma unroll
    for (int i = 0; i < 3; ++i)
        #pragma unroll
        for (int j = 0; j < 3; ++j) {
            float s = 0.f;
            #pragma unroll
            for (int k = 0; k < 3; ++k) s += iK1[k*3+i] * T[k*3+j];  // iK1^T
            F[i*3+j] = s;
        }

    float a = F[0]*x + F[1]*y + F[2];
    float b = F[3]*x + F[4]*y + F[5];
    float c = F[6]*x + F[7]*y + F[8];
    float nrm = fmaxf(sqrtf(a*a + b*b), 1e-12f);
    a /= nrm; b /= nrm; c /= nrm;
    bool mode = fabsf(b) >= fabsf(a);
    float den = mode ? b : a;
    if (fabsf(den) < 1e-12f) den = 1e-12f;
    float p = (mode ? -a : -b) / den;
    float q = -c / den;
    lines[idx] = make_float4(p, q, mode ? 1.f : 0.f, 0.f);
}

// Masked dense linear attention. One wave32 per (n, h, 16-row l-tile), 32 keys/iter.
//   sc0,sc1 = phiQ(16x16) . phiK^T(16x32)   -> 2x v_wmma_f32_16x16x32_bf16 (d padded to 32)
//   num    += mask(scores)(16x32) . V(32x16) -> 1x v_wmma, full K=32 contraction
//   den[l]  = running row-sum of masked scores, shfl_xor reduced at the end.
__global__ __launch_bounds__(128) void attn_kernel(
        const __bf16* __restrict__ Qb,
        const __bf16* __restrict__ Kb,
        const __bf16* __restrict__ Vt,
        const unsigned char* __restrict__ solved,
        const float4* __restrict__ lines,
        const float2* __restrict__ coords,
        float* __restrict__ out,
        int N, int L) {
    const int H = NHEAD, D = DIM;
    __shared__ __attribute__((aligned(16))) float sbuf[4][16 * 36]; // 16x32 tile, stride 36

    const int wave = threadIdx.x >> 5;
    const int lane = threadIdx.x & 31;
    const int row  = lane & 15;          // A-row / B,C-column index
    const int half = lane >> 4;

    const int tiles = (L + 15) >> 4;
    const int tile = blockIdx.x * 4 + wave;
    if (tile >= tiles) return;
    const int h = blockIdx.y;
    const int n = blockIdx.z;
    const int l0 = tile << 4;
    const bool sol = solved[n] != 0;

    // Line params for the 8 score rows this lane masks: M = r + 8*half
    float lp[8], lq[8], lm[8];
    if (sol) {
        #pragma unroll
        for (int r = 0; r < 8; ++r) {
            int li = l0 + r + 8*half; if (li >= L) li = L - 1;
            float4 ln = lines[(size_t)n * L + li];
            lp[r] = ln.x; lq[r] = ln.y; lm[r] = ln.z;
        }
    }

    // A operand: phi(Q). Lane = row l0+row, half 0 elements 0..15 = d 0..15, half 1 zero.
    v16bf aQ;
    {
        int li = l0 + row; if (li >= L) li = L - 1;
        const __bf16* qrow = Qb + (((size_t)n * L + li) * H + h) * D;
        aQ = (half == 0) ? *(const v16bf*)qrow : zero16bf();   // 32B load
    }

    v8f num;
    float denAcc[8];
    #pragma unroll
    for (int r = 0; r < 8; ++r) { num[r] = 0.0f; denAcc[r] = 0.0f; }

    float* sb = sbuf[wave];
    const __bf16* vtrow = Vt + (((size_t)n * H + h) * D + row) * (size_t)L;

    for (int s0 = 0; s0 < L; s0 += 32) {
        // ---- B operands for scores: phi(K)^T, d-dim on low half-wave lanes only
        v16bf bK0 = zero16bf();
        v16bf bK1 = zero16bf();
        if (half == 0) {
            int si0 = s0 + row;      if (si0 >= L) si0 = L - 1;
            int si1 = s0 + 16 + row; if (si1 >= L) si1 = L - 1;
            bK0 = *(const v16bf*)(Kb + (((size_t)n * L + si0) * H + h) * D); // 32B
            bK1 = *(const v16bf*)(Kb + (((size_t)n * L + si1) * H + h) * D); // 32B
            if (s0 + 32 < L)
                __builtin_prefetch(Kb + (((size_t)n * L + s0 + 32 + row) * H + h) * D, 0, 3);
        }

        v8f sc0, sc1;
        #pragma unroll
        for (int r = 0; r < 8; ++r) { sc0[r] = 0.0f; sc1[r] = 0.0f; }
        sc0 = __builtin_amdgcn_wmma_f32_16x16x32_bf16(false, aQ, false, bK0,
                                                      (short)0, sc0, false, false);
        sc1 = __builtin_amdgcn_wmma_f32_16x16x32_bf16(false, aQ, false, bK1,
                                                      (short)0, sc1, false, false);

        // ---- epipolar mask + denominator + stage 16x32 tile to LDS
        #pragma unroll
        for (int t = 0; t < 2; ++t) {
            const int s = s0 + 16*t + row;           // this lane's key column
            const int sidx = (s < L) ? s : (L - 1);
            const float2 cxy = coords[sidx];
            const float sx = cxy.x, sy = cxy.y;
            const bool colok = sol ? ((s < L) && (s != 0)) : (s < L);
            #pragma unroll
            for (int r = 0; r < 8; ++r) {
                float v = t ? sc1[r] : sc0[r];
                bool w = colok;
                if (sol) {
                    float par  = (lm[r] != 0.0f) ? sx : sy;
                    float perp = (lm[r] != 0.0f) ? sy : sx;
                    float cc = fmaf(lp[r], par, lq[r]);
                    w = w && (fabsf(perp - cc) < 1.5f);
                }
                v = w ? v : 0.0f;
                denAcc[r] += v;
                sb[(r + 8*half) * 36 + 16*t + row] = v;
            }
        }
        // cross-lane LDS visibility within the wave (CDNA5 split DS counter)
        asm volatile("s_wait_dscnt 0x0" ::: "memory");

        // ---- A = masked scores in A layout: elems 0..15 = K 16h..16h+15
        v16bf aS;
        {
            const float* srow = sb + row * 36 + 16*half;
            float4 f0 = *(const float4*)(srow);
            float4 f1 = *(const float4*)(srow + 4);
            float4 f2 = *(const float4*)(srow + 8);
            float4 f3 = *(const float4*)(srow + 12);
            aS[0]  = (__bf16)f0.x;  aS[1]  = (__bf16)f0.y;
            aS[2]  = (__bf16)f0.z;  aS[3]  = (__bf16)f0.w;
            aS[4]  = (__bf16)f1.x;  aS[5]  = (__bf16)f1.y;
            aS[6]  = (__bf16)f1.z;  aS[7]  = (__bf16)f1.w;
            aS[8]  = (__bf16)f2.x;  aS[9]  = (__bf16)f2.y;
            aS[10] = (__bf16)f2.z;  aS[11] = (__bf16)f2.w;
            aS[12] = (__bf16)f3.x;  aS[13] = (__bf16)f3.y;
            aS[14] = (__bf16)f3.z;  aS[15] = (__bf16)f3.w;
        }

        // ---- B = V chunk from transposed bf16 V: lane col v=row, K = s0+16*half+j
        v16bf bV = *(const v16bf*)(vtrow + s0 + 16*half);   // 32B load
        if (s0 + 32 < L)
            __builtin_prefetch(vtrow + s0 + 32 + 16*half, 0, 3);

        num = __builtin_amdgcn_wmma_f32_16x16x32_bf16(false, aS, false, bV,
                                                      (short)0, num, false, false);
    }

    // Reduce denominator across the 16 lanes of each half (rows M = r + 8*half)
    #pragma unroll
    for (int r = 0; r < 8; ++r) {
        float d = denAcc[r];
        d += __shfl_xor(d, 1, 32);
        d += __shfl_xor(d, 2, 32);
        d += __shfl_xor(d, 4, 32);
        d += __shfl_xor(d, 8, 32);
        denAcc[r] = d;
    }

    // Write out: element (M = r + 8*half, v = lane&15)
    #pragma unroll
    for (int r = 0; r < 8; ++r) {
        int li = l0 + 8*half + r;
        if (li < L) {
            float o = num[r] / (denAcc[r] + EPS_F);
            out[(((size_t)n * L + li) * H + h) * D + row] = o;
        }
    }
}

extern "C" void kernel_launch(void* const* d_in, const int* in_sizes, int n_in,
                              void* d_out, int out_size, void* d_ws, size_t ws_size,
                              hipStream_t stream) {
    const float* Q  = (const float*)d_in[0];
    const float* K  = (const float*)d_in[1];
    const float* V  = (const float*)d_in[2];
    const unsigned char* solved = (const unsigned char*)d_in[3];
    const float* K0 = (const float*)d_in[4];
    const float* K1 = (const float*)d_in[5];
    const float* R  = (const float*)d_in[6];
    const float* t  = (const float*)d_in[7];
    const int* w0c  = (const int*)d_in[9];
    const int* scale = (const int*)d_in[10];

    const int N = in_sizes[4] / 9;                      // K0 is N x 3 x 3
    const int L = in_sizes[0] / (N * NHEAD * DIM);      // queries: N x L x H x D

    // Workspace: phiQ | phiK | V^T (bf16, each N*L*H*D) + pad | lines | coords
    const size_t NB = (size_t)N * L * NHEAD * DIM;
    __bf16* Qb = (__bf16*)d_ws;
    __bf16* Kb = Qb + NB;
    __bf16* Vt = Kb + NB;
    float4* lines = (float4*)(Vt + NB + 64);            // 64-elem pad for tail vector loads
    float2* coords = (float2*)(lines + (size_t)N * L);

    size_t total = NB;
    prep_kernel<<<(unsigned)((total + 255) / 256), 256, 0, stream>>>(
        Q, K, V, Qb, Kb, Vt, N, L);

    int totalNL = N * L;
    epi_lines_kernel<<<(totalNL + 255) / 256, 256, 0, stream>>>(
        K0, K1, R, t, w0c, scale, N, L, lines, coords);

    int tiles = (L + 15) / 16;
    dim3 grid((tiles + 3) / 4, NHEAD, N);
    attn_kernel<<<grid, 128, 0, stream>>>(
        Qb, Kb, Vt, solved, lines, coords, (float*)d_out, N, L);
}
